// Head_15393162788975
// MI455X (gfx1250) — compile-verified, matrix operand order checked by reading, not run on hardware
//
#include <hip/hip_runtime.h>
#include <hip/hip_bf16.h>
#include <math.h>

// ---------- fixed problem dims (reference: B=16, T=E=H=1024) ----------
constexpr int Bc = 16;
constexpr int Tc = 1024;
constexpr int Ec = 1024;
constexpr int Hc = 1024;
constexpr int PSTR = 1032;     // LDS panel row stride in halves (1024 + 8 pad)

// ---------- CDNA5 WMMA types ----------
typedef __attribute__((ext_vector_type(16))) __bf16 v16bf;
typedef __attribute__((ext_vector_type(8)))  __bf16 v8bf;
typedef __attribute__((ext_vector_type(8)))  float  v8f;

__device__ __forceinline__ __bf16 f32_to_bf16(float f) {
  return static_cast<__bf16>(f);          // native v_cvt_pk_bf16_f32 on gfx1250
}

__device__ __forceinline__ v8f wmma_bf16(v16bf a, v16bf b, v8f c) {
  // (neg_a, A, neg_b, B, c_mod, C, reuse_a, reuse_b)
  return __builtin_amdgcn_wmma_f32_16x16x32_bf16(false, a, false, b, (short)0, c,
                                                 false, false);
}

// Fragment per ISA 16-bit A layout: lanes 0-15 hold K=kk+0..7 / kk+16..23,
// lanes 16-31 hold K=kk+8..15 / kk+24..31.  p0/p1 are 16B-aligned.
__device__ __forceinline__ v16bf load_frag_bf16(const __bf16* p0, const __bf16* p1) {
  v8bf lo = *(const v8bf*)p0;
  v8bf hi = *(const v8bf*)p1;
  v16bf f;
#pragma unroll
  for (int i = 0; i < 8; ++i) { f[i] = lo[i]; f[8 + i] = hi[i]; }
  return f;
}

__device__ __forceinline__ v16bf load_frag_f32(const float* p0, const float* p1) {
  float4 a0 = ((const float4*)p0)[0];
  float4 a1 = ((const float4*)p0)[1];
  float4 b0 = ((const float4*)p1)[0];
  float4 b1 = ((const float4*)p1)[1];
  v16bf f;
  f[0]  = f32_to_bf16(a0.x); f[1]  = f32_to_bf16(a0.y);
  f[2]  = f32_to_bf16(a0.z); f[3]  = f32_to_bf16(a0.w);
  f[4]  = f32_to_bf16(a1.x); f[5]  = f32_to_bf16(a1.y);
  f[6]  = f32_to_bf16(a1.z); f[7]  = f32_to_bf16(a1.w);
  f[8]  = f32_to_bf16(b0.x); f[9]  = f32_to_bf16(b0.y);
  f[10] = f32_to_bf16(b0.z); f[11] = f32_to_bf16(b0.w);
  f[12] = f32_to_bf16(b1.x); f[13] = f32_to_bf16(b1.y);
  f[14] = f32_to_bf16(b1.z); f[15] = f32_to_bf16(b1.w);
  return f;
}

// ---------- CDNA5 async global->LDS copy (ASYNCcnt path) ----------
__device__ __forceinline__ unsigned lds_off(const void* p) {
  // flat shared pointer: addr[31:0] is the LDS byte offset
  return (unsigned)(uintptr_t)p;
}

__device__ __forceinline__ void async_copy_b128(unsigned dst, const void* src) {
  asm volatile("global_load_async_to_lds_b128 %0, %1, off"
               :: "v"(dst), "v"(src) : "memory");
}

__device__ __forceinline__ void async_wait0() {
  asm volatile("s_wait_asynccnt 0" ::: "memory");
}

// Stage a 16 x 1024 bf16 row-major panel (row stride 1024) into LDS with
// padded stride PSTR.  256 threads: thread t copies 128B of row t>>4.
__device__ __forceinline__ void stage_panel_async(const __bf16* __restrict__ g,
                                                  __bf16* sh, int tid) {
  const int row = tid >> 4;
  const int cb  = (tid & 15) * 128;                       // byte offset in row
  const char* src = (const char*)(g + (size_t)row * 1024) + cb;
  const unsigned dst = lds_off(sh) + (unsigned)(row * (PSTR * 2) + cb);
#pragma unroll
  for (int i = 0; i < 8; ++i)
    async_copy_b128(dst + i * 16, src + i * 16);
}

// ---------------------------------------------------------------------------
// Stage 1: Y[m,n] = sum_k X[m,k] * W[n,k]  (X:[M,K]=[B*T,E] f32, W:[N,K]=[H,E])
// TRANSPOSE=true stores per-batch transposed Y[b, n, t]  (used for V).
// Grid: (H/128, B*T/16), 256 threads (8 waves, one 16x16 tile each).
// ---------------------------------------------------------------------------
template <bool TRANSPOSE>
__global__ __launch_bounds__(256) void proj_kernel(
    const float* __restrict__ X, const float* __restrict__ W,
    __bf16* __restrict__ Y, float scale)
{
  const int lane  = threadIdx.x & 31;
  const int wave  = threadIdx.x >> 5;
  const int m0    = blockIdx.y << 4;
  const int n0    = (blockIdx.x * 8 + wave) << 4;
  const int r     = lane & 15;
  const int khalf = (lane >> 4) << 3;

  const float* arow = X + (size_t)(m0 + r) * Ec;
  const float* brow = W + (size_t)(n0 + r) * Ec;

  v8f acc = {};
  for (int kk = 0; kk < Ec; kk += 32) {
    if (kk + 32 < Ec) {                       // gfx1250 global_prefetch path
      __builtin_prefetch(arow + kk + 32 + khalf, 0, 1);
      __builtin_prefetch(brow + kk + 32 + khalf, 0, 1);
    }
    v16bf a = load_frag_f32(arow + kk + khalf, arow + kk + 16 + khalf);
    v16bf b = load_frag_f32(brow + kk + khalf, brow + kk + 16 + khalf);
    acc = wmma_bf16(a, b, acc);
  }

#pragma unroll
  for (int rr = 0; rr < 8; ++rr) {
    int row = m0 + ((lane < 16) ? rr : rr + 8);
    int col = n0 + (lane & 15);
    float v = acc[rr] * scale;
    if constexpr (!TRANSPOSE) {
      Y[(size_t)row * Hc + col] = f32_to_bf16(v);
    } else {
      int bb = row >> 10, s = row & (Tc - 1);          // [B, H, T]
      Y[((size_t)bb * Hc + col) * Tc + s] = f32_to_bf16(v);
    }
  }
}

// ---------------------------------------------------------------------------
// Stage 2: S = QK^T (K-dim = H), causal (s <= t+1) + padding mask, softmax
// over s, P written bf16.  One block per (b, 16 query rows); Q panel staged
// in LDS via async copy; 8 waves cover the 64 key tiles; scores in LDS.
// ---------------------------------------------------------------------------
__global__ __launch_bounds__(256) void attn_scores_softmax_kernel(
    const __bf16* __restrict__ Q, const __bf16* __restrict__ Kb,
    const int* __restrict__ pad, __bf16* __restrict__ P)
{
  __shared__ float  S[16 * Tc];          // 64 KB score rows
  __shared__ __bf16 qsh[16 * PSTR];      // 33 KB staged Q panel
  __shared__ float  red[16][16];
  __shared__ float  rowstat[16];

  const int tid   = threadIdx.x;
  const int b     = blockIdx.y;
  const int t0    = blockIdx.x << 4;
  const int lane  = tid & 31;
  const int wave  = tid >> 5;
  const int r     = lane & 15;
  const int khalf = (lane >> 4) << 3;

  stage_panel_async(Q + ((size_t)b * Tc + t0) * Hc, qsh, tid);
  async_wait0();
  __syncthreads();

  const __bf16* qbase = &qsh[r * PSTR];

  for (int ct = wave; ct < (Tc >> 4); ct += 8) {
    const int s0 = ct << 4;
    const __bf16* krow = Kb + ((size_t)b * Tc + s0 + r) * Hc;
    v8f acc = {};
    for (int kk = 0; kk < Hc; kk += 32) {
      if (kk + 32 < Hc) __builtin_prefetch(krow + kk + 32 + khalf, 0, 1);
      v16bf a  = load_frag_bf16(qbase + kk + khalf, qbase + kk + 16 + khalf);
      v16bf bb = load_frag_bf16(krow + kk + khalf, krow + kk + 16 + khalf);
      acc = wmma_bf16(a, bb, acc);
    }
#pragma unroll
    for (int rr = 0; rr < 8; ++rr) {
      int lrow = (lane < 16) ? rr : rr + 8;
      int t = t0 + lrow;
      int s = s0 + (lane & 15);
      bool ok = (s <= t + 1) && (pad[b * Tc + s] == 0);
      S[lrow * Tc + s] = ok ? acc[rr] : -INFINITY;
    }
  }
  __syncthreads();

  // softmax: 16 threads per row
  const int rw = tid >> 4;
  const int g  = tid & 15;

  float m = -INFINITY;
  for (int i = g; i < Tc; i += 16) m = fmaxf(m, S[rw * Tc + i]);
  red[rw][g] = m;
  __syncthreads();
  if (g == 0) {
    float mm = red[rw][0];
#pragma unroll
    for (int i = 1; i < 16; ++i) mm = fmaxf(mm, red[rw][i]);
    rowstat[rw] = mm;
  }
  __syncthreads();
  const float rmax = rowstat[rw];

  float sum = 0.f;
  for (int i = g; i < Tc; i += 16) {
    float e = __expf(S[rw * Tc + i] - rmax);
    S[rw * Tc + i] = e;
    sum += e;
  }
  __syncthreads();
  red[rw][g] = sum;
  __syncthreads();
  if (g == 0) {
    float ss = red[rw][0];
#pragma unroll
    for (int i = 1; i < 16; ++i) ss += red[rw][i];
    rowstat[rw] = 1.0f / ss;
  }
  __syncthreads();
  const float inv = rowstat[rw];

  __bf16* prow = P + ((size_t)b * Tc + t0 + rw) * Tc;
  for (int i = g; i < Tc; i += 16) prow[i] = f32_to_bf16(S[rw * Tc + i] * inv);
}

// ---------------------------------------------------------------------------
// Stage 3: O[b,t,h] = sum_s P[b,t,s] * Vt[b,h,s]   (bf16 in, f32 out)
// P panel (shared by all 8 waves) staged in LDS via async copy.
// Grid: (H/128, T/16, B), 256 threads.
// ---------------------------------------------------------------------------
__global__ __launch_bounds__(256) void pv_kernel(
    const __bf16* __restrict__ P, const __bf16* __restrict__ Vt,
    float* __restrict__ O)
{
  __shared__ __bf16 ash[16 * PSTR];      // 33 KB staged P panel

  const int tid   = threadIdx.x;
  const int lane  = tid & 31;
  const int wave  = tid >> 5;
  const int b     = blockIdx.z;
  const int m0    = blockIdx.y << 4;
  const int n0    = (blockIdx.x * 8 + wave) << 4;
  const int r     = lane & 15;
  const int khalf = (lane >> 4) << 3;

  stage_panel_async(P + ((size_t)b * Tc + m0) * Tc, ash, tid);
  async_wait0();
  __syncthreads();

  const __bf16* arow = &ash[r * PSTR];
  const __bf16* brow = Vt + ((size_t)b * Hc + n0 + r) * Tc;

  v8f acc = {};
  for (int kk = 0; kk < Tc; kk += 32) {
    if (kk + 32 < Tc) __builtin_prefetch(brow + kk + 32 + khalf, 0, 1);
    v16bf a  = load_frag_bf16(arow + kk + khalf, arow + kk + 16 + khalf);
    v16bf bb = load_frag_bf16(brow + kk + khalf, brow + kk + 16 + khalf);
    acc = wmma_bf16(a, bb, acc);
  }

#pragma unroll
  for (int rr = 0; rr < 8; ++rr) {
    int row = m0 + ((lane < 16) ? rr : rr + 8);
    int col = n0 + (lane & 15);
    O[((size_t)b * Tc + row) * Hc + col] = acc[rr];
  }
}

// ---------------------------------------------------------------------------
extern "C" void kernel_launch(void* const* d_in, const int* in_sizes, int n_in,
                              void* d_out, int out_size, void* d_ws, size_t ws_size,
                              hipStream_t stream) {
  const float* key   = (const float*)d_in[0];
  const float* query = (const float*)d_in[1];
  const float* value = (const float*)d_in[2];
  const int*   pad   = (const int*)d_in[3];
  const float* Wk    = (const float*)d_in[4];
  const float* Wq    = (const float*)d_in[5];
  const float* Wv    = (const float*)d_in[6];
  float* out = (float*)d_out;

  // workspace: q,k bf16 [B,T,H]; v transposed bf16 [B,H,T]; P bf16 [B,T,T]
  const size_t nElem = (size_t)Bc * Tc * Hc;
  __bf16* qb = (__bf16*)d_ws;
  __bf16* kb = qb + nElem;
  __bf16* vt = kb + nElem;
  __bf16* Pm = vt + nElem;          // total 4 * nElem * 2 = 128 MB

  const float scale = 1.0f / sqrtf((float)Ec);

  dim3 blk(256);
  dim3 gproj(Hc / 128, (Bc * Tc) / 16);
  proj_kernel<false><<<gproj, blk, 0, stream>>>(query, Wq, qb, scale);
  proj_kernel<false><<<gproj, blk, 0, stream>>>(key,   Wk, kb, 1.0f);
  proj_kernel<true ><<<gproj, blk, 0, stream>>>(value, Wv, vt, 1.0f);

  dim3 gscore(Tc / 16, Bc);
  attn_scores_softmax_kernel<<<gscore, blk, 0, stream>>>(qb, kb, pad, Pm);

  dim3 gpv(Hc / 128, Tc / 16, Bc);
  pv_kernel<<<gpv, blk, 0, stream>>>(Pm, vt, out);
}
